// GATConvModel_20057497272867
// MI455X (gfx1250) — matrix operand order repaired and checked by reading, not compiled, not run on hardware
//
#include <hip/hip_runtime.h>
#include <hip/hip_bf16.h>
#include <stdint.h>

// ---------------------------------------------------------------------------
// CDNA5 (gfx1250) GAT forward. GEMMs run on bf16 WMMA (v_wmma_f32_16x16x32_bf16)
// with f32 accumulation; operands are pre-swizzled into the exact per-lane
// fragment layouts from CDNA5 ISA so every fragment is one
// contiguous 32B load per lane.
// ---------------------------------------------------------------------------

typedef __bf16 bf16_t;
typedef __attribute__((ext_vector_type(16))) __bf16 v16bf;
typedef __attribute__((ext_vector_type(8)))  float  v8f;

#define HDIM 1024
#define NGRP 16   // 64 N-tiles / 4 tiles per wave

// ---------------- conversion: f32 weight [K,1024] -> swizzled bf16 B tiles ---
// tile t = nt*KT + kt holds 512 bf16; within tile: offset = lane*16 + e with
// K = kt*32 + (lane/16)*8 + (e<8 ? e : e+8), N = nt*16 + lane%16   (B 32x16 fragment layout)
__global__ __launch_bounds__(256) void k_conv_w(const float* __restrict__ W,
                                                bf16_t* __restrict__ out,
                                                int Kreal, int KT) {
  size_t idx = (size_t)blockIdx.x * 256 + threadIdx.x;
  size_t tot = (size_t)64 * KT * 512;
  if (idx >= tot) return;
  int e = (int)(idx & 15);
  int L = (int)((idx >> 4) & 31);
  size_t t = idx >> 9;
  int kt = (int)(t % KT);
  int nt = (int)(t / KT);
  int k = kt * 32 + ((L >> 4) << 3) + ((e < 8) ? e : e + 8);
  int n = (nt << 4) + (L & 15);
  float v = (k < Kreal) ? W[(size_t)k * HDIM + n] : 0.f;
  out[idx] = (bf16_t)v;
}

// ---------------- conversion: f32 activation [M,ld] -> swizzled bf16 A tiles -
// tile t = mt*KT + kt; lane L row = mt*16 + L%16;
// K = kt*32 + (L/16)*8 + (e<8 ? e : e+8)              (A 16x32 fragment layout)
__global__ __launch_bounds__(256) void k_conv_act(const float* __restrict__ X,
                                                  bf16_t* __restrict__ out,
                                                  int Kreal, int ld, int KT,
                                                  size_t tot) {
  size_t idx = (size_t)blockIdx.x * 256 + threadIdx.x;
  if (idx >= tot) return;
  int e = (int)(idx & 15);
  int L = (int)((idx >> 4) & 31);
  size_t t = idx >> 9;
  int kt = (int)(t % KT);
  int mt = (int)(t / KT);
  int k = kt * 32 + ((L >> 4) << 3) + ((e < 8) ? e : e + 8);
  int row = (mt << 4) + (L & 15);
  float v = (k < Kreal) ? X[(size_t)row * ld + k] : 0.f;
  out[idx] = (bf16_t)v;
}

// ---------------- bf16 WMMA GEMM: C[M,1024] = A[M,K] * B[K,1024] -------------
// One wave per 16x64 output strip (4 N-tiles share the A fragment).
__global__ __launch_bounds__(256) void k_gemm(const bf16_t* __restrict__ A,
                                              const bf16_t* __restrict__ Bsw,
                                              float* __restrict__ C,
                                              int Mt, int KT) {
  int lane = threadIdx.x & 31;
  int wave = blockIdx.x * 8 + (threadIdx.x >> 5);
  if (wave >= Mt * NGRP) return;            // wave-uniform; EXEC stays full
  int mt = wave / NGRP;
  int ng = wave % NGRP;
  const v16bf* Ap = (const v16bf*)(A   + ((size_t)mt * KT * 32 + lane) * 16);
  const v16bf* Bp = (const v16bf*)(Bsw + ((size_t)ng * 4 * KT * 32 + lane) * 16);
  v8f c0 = {}, c1 = {}, c2 = {}, c3 = {};
  for (int kt = 0; kt < KT; ++kt) {
    v16bf a  = Ap[(size_t)kt * 32];
    v16bf b0 = Bp[(size_t)kt * 32];
    v16bf b1 = Bp[(size_t)(KT + kt) * 32];
    v16bf b2 = Bp[(size_t)(2 * KT + kt) * 32];
    v16bf b3 = Bp[(size_t)(3 * KT + kt) * 32];
    c0 = __builtin_amdgcn_wmma_f32_16x16x32_bf16(false, a, false, b0, (short)0, c0, false, false);
    c1 = __builtin_amdgcn_wmma_f32_16x16x32_bf16(false, a, false, b1, (short)0, c1, false, false);
    c2 = __builtin_amdgcn_wmma_f32_16x16x32_bf16(false, a, false, b2, (short)0, c2, false, false);
    c3 = __builtin_amdgcn_wmma_f32_16x16x32_bf16(false, a, false, b3, (short)0, c3, false, false);
  }
  // C fragment: VGPR g -> row (lane/16)*8+g, col lane%16
  int rb = (mt << 4) + ((lane >> 4) << 3);
  int cb = (ng << 6) + (lane & 15);
#pragma unroll
  for (int g = 0; g < 8; ++g) {
    float* r = C + (size_t)(rb + g) * HDIM + cb;
    r[0] = c0[g]; r[16] = c1[g]; r[32] = c2[g]; r[48] = c3[g];
  }
}

// ---------------- el/er head dots: out[n*8+h] = <X[n, h*128: ], av[h*128: ]> -
__global__ __launch_bounds__(256) void k_rowdot(const float* __restrict__ X,
                                                const float* __restrict__ av,
                                                float* __restrict__ out, int NH) {
  int lane = threadIdx.x & 31;
  int w = blockIdx.x * 8 + (threadIdx.x >> 5);
  if (w >= NH) return;
  int n = w >> 3, h = w & 7;
  const float* x = X + (size_t)n * HDIM + h * 128;
  const float* a = av + h * 128;
  float s = 0.f;
#pragma unroll
  for (int i = 0; i < 4; ++i) s += x[lane + 32 * i] * a[lane + 32 * i];
  for (int o = 16; o; o >>= 1) s += __shfl_down(s, o, 32);
  if (!lane) out[w] = s;
}

__global__ void k_fill(float* p, float v, size_t n) {
  size_t i = (size_t)blockIdx.x * 256 + threadIdx.x;
  if (i < n) p[i] = v;
}

__device__ inline void atomicMaxF(float* addr, float v) {
  if (v >= 0.f) atomicMax((int*)addr, __float_as_int(v));
  else          atomicMin((unsigned int*)addr, __float_as_uint(v));
}

__global__ void k_edge_logits(const float* __restrict__ el, const float* __restrict__ er,
                              const int* __restrict__ src, const int* __restrict__ dst,
                              float* __restrict__ eb, float* __restrict__ m, int E8) {
  int i = blockIdx.x * 256 + threadIdx.x;
  if (i >= E8) return;
  int e = i >> 3, h = i & 7;
  float x = el[src[e] * 8 + h] + er[dst[e] * 8 + h];
  x = x > 0.f ? x : 0.2f * x;             // leaky_relu(0.2)
  eb[i] = x;
  atomicMaxF(&m[dst[e] * 8 + h], x);
}

__global__ void k_edge_exp(float* __restrict__ eb, const float* __restrict__ m,
                           float* __restrict__ den, const int* __restrict__ dst, int E8) {
  int i = blockIdx.x * 256 + threadIdx.x;
  if (i >= E8) return;
  int e = i >> 3, h = i & 7;
  float x = expf(eb[i] - m[dst[e] * 8 + h]);
  eb[i] = x;
  atomicAdd(&den[dst[e] * 8 + h], x);
}

// one block per edge; each thread scatters 4 contiguous channels
__global__ __launch_bounds__(256) void k_edge_scatter(const float* __restrict__ fs,
                                                      const float* __restrict__ eb,
                                                      const float* __restrict__ den,
                                                      const int* __restrict__ src,
                                                      const int* __restrict__ dst,
                                                      float* __restrict__ rst) {
  int e = blockIdx.x;
  int s = src[e], d = dst[e];
  int c = threadIdx.x * 4;
  int h = c >> 7;
  float a = eb[e * 8 + h] / den[d * 8 + h];
  const float4 f = *(const float4*)(fs + (size_t)s * HDIM + c);
  float* r = rst + (size_t)d * HDIM + c;
  atomicAdd(r + 0, f.x * a); atomicAdd(r + 1, f.y * a);
  atomicAdd(r + 2, f.z * a); atomicAdd(r + 3, f.w * a);
}

// fused: x = rst + res + bias ; LayerNorm(1024) ; ELU ; one block per node row
__global__ __launch_bounds__(256) void k_comb_ln(const float* __restrict__ rst,
                                                 const float* __restrict__ res,
                                                 const float* __restrict__ bias,
                                                 const float* __restrict__ g,
                                                 const float* __restrict__ b,
                                                 float* __restrict__ out) {
  int n = blockIdx.x;
  __shared__ float sh1[8], sh2[8], bc[2];
  float x[4]; float s1 = 0.f, s2 = 0.f;
#pragma unroll
  for (int i = 0; i < 4; ++i) {
    int c = i * 256 + threadIdx.x;
    float v = rst[(size_t)n * HDIM + c] + res[(size_t)n * HDIM + c] + bias[c];
    x[i] = v; s1 += v; s2 += v * v;
  }
  int lane = threadIdx.x & 31, w = threadIdx.x >> 5;
  for (int o = 16; o; o >>= 1) { s1 += __shfl_down(s1, o, 32); s2 += __shfl_down(s2, o, 32); }
  if (!lane) { sh1[w] = s1; sh2[w] = s2; }
  __syncthreads();
  if (threadIdx.x == 0) {
    float t1 = 0.f, t2 = 0.f;
    for (int i = 0; i < 8; ++i) { t1 += sh1[i]; t2 += sh2[i]; }
    float mu = t1 * (1.f / HDIM);
    float var = t2 * (1.f / HDIM) - mu * mu;
    bc[0] = mu; bc[1] = rsqrtf(var + 1e-5f);
  }
  __syncthreads();
  float mu = bc[0], rstd = bc[1];
#pragma unroll
  for (int i = 0; i < 4; ++i) {
    int c = i * 256 + threadIdx.x;
    float y = (x[i] - mu) * rstd * g[c] + b[c];
    y = y > 0.f ? y : (expf(y) - 1.f);    // ELU
    out[(size_t)n * HDIM + c] = y;
  }
}

// layer-2: mean over heads, then LayerNorm(128) + ELU; one block (128 thr)/node
__global__ __launch_bounds__(128) void k_comb_mean_ln(const float* __restrict__ rst,
                                                      const float* __restrict__ res,
                                                      const float* __restrict__ bias,
                                                      const float* __restrict__ g,
                                                      const float* __restrict__ b,
                                                      float* __restrict__ out) {
  int n = blockIdx.x;
  int d = threadIdx.x;
  __shared__ float sh1[4], sh2[4], bc[2];
  float v = 0.f;
#pragma unroll
  for (int h = 0; h < 8; ++h) {
    size_t o = (size_t)n * HDIM + h * 128 + d;
    v += rst[o] + res[o] + bias[h * 128 + d];
  }
  v *= 0.125f;
  float s1 = v, s2 = v * v;
  int lane = threadIdx.x & 31, w = threadIdx.x >> 5;
  for (int o = 16; o; o >>= 1) { s1 += __shfl_down(s1, o, 32); s2 += __shfl_down(s2, o, 32); }
  if (!lane) { sh1[w] = s1; sh2[w] = s2; }
  __syncthreads();
  if (threadIdx.x == 0) {
    float t1 = 0.f, t2 = 0.f;
    for (int i = 0; i < 4; ++i) { t1 += sh1[i]; t2 += sh2[i]; }
    float mu = t1 * (1.f / 128.f);
    float var = t2 * (1.f / 128.f) - mu * mu;
    bc[0] = mu; bc[1] = rsqrtf(var + 1e-5f);
  }
  __syncthreads();
  float y = (v - bc[0]) * bc[1] * g[d] + b[d];
  y = y > 0.f ? y : (expf(y) - 1.f);
  out[(size_t)n * 128 + d] = y;
}

__global__ void k_graph_accum(const float* __restrict__ h, const int* __restrict__ gid,
                              float* __restrict__ gsum, float* __restrict__ gcnt, int N) {
  int i = blockIdx.x * 256 + threadIdx.x;
  if (i >= N * 128) return;
  int n = i >> 7, d = i & 127;
  atomicAdd(&gsum[gid[n] * 128 + d], h[i]);
  if (!d) atomicAdd(&gcnt[gid[n]], 1.f);
}

__global__ __launch_bounds__(256) void k_head(const float* gsL, const float* gcL,
                                              const float* gsN, const float* gcN,
                                              const float* W1, const float* b1,
                                              const float* Wo, const float* bo,
                                              float* out) {
  __shared__ float hg[8 * 128];
  __shared__ float tb[8 * 128];
  for (int i = threadIdx.x; i < 1024; i += 256) {
    int b = i >> 7;
    hg[i] = gsL[i] / fmaxf(gcL[b], 1.f) + gsN[i] / fmaxf(gcN[b], 1.f);
  }
  __syncthreads();
  for (int i = threadIdx.x; i < 1024; i += 256) {
    int b = i >> 7, j = i & 127;
    float s = b1[j];
    for (int d = 0; d < 128; ++d) s += hg[b * 128 + d] * W1[d * 128 + j];
    tb[i] = fmaxf(s, 0.f) * Wo[j];
  }
  __syncthreads();
  if (threadIdx.x < 8) {
    float s = bo[0];
    for (int j = 0; j < 128; ++j) s += tb[threadIdx.x * 128 + j];
    out[threadIdx.x] = s;
  }
}

// ---------------------------------------------------------------------------
extern "C" void kernel_launch(void* const* d_in, const int* in_sizes, int n_in,
                              void* d_out, int out_size, void* d_ws, size_t ws_size,
                              hipStream_t stream) {
  (void)n_in; (void)out_size; (void)ws_size;
  const float* feat_l  = (const float*)d_in[0];
  const float* feat_n  = (const float*)d_in[1];
  const int* l2n_src   = (const int*)d_in[2];
  const int* l2n_dst   = (const int*)d_in[3];
  const int* n2l_src   = (const int*)d_in[4];
  const int* n2l_dst   = (const int*)d_in[5];
  const int* gid_l     = (const int*)d_in[6];
  const int* gid_n     = (const int*)d_in[7];

  const int NL = in_sizes[6], NN = in_sizes[7];
  const int E  = in_sizes[2];
  const int KL = in_sizes[0] / NL;          // 300
  const int KN = in_sizes[1] / NN;          // 200
  const int KT_L = (KL + 31) / 32, KT_N = (KN + 31) / 32, KT_H = HDIM / 32;
  const int MtL = NL / 16, MtN = NN / 16;   // 625 each

  // params: JAX pytree flatten order (dict keys sorted; 'W'<'a'<'b'<'r')
  const float* P[64];
  for (int i = 8; i < 52; ++i) P[i] = (const float*)d_in[i];
  // conv0.l2n: 8 Wdst, 9 Wsrc, 10 al, 11 ar, 12 bias, 13 res
  // conv0.n2l: 14 Wdst, 15 Wsrc, 16 al, 17 ar, 18 bias, 19 res
  // conv1.l2n: 20 W, 21 al, 22 ar, 23 bias ; conv1.n2l: 24..27
  // conv2.l2n: 28..31 ; conv2.n2l: 32..35
  // fc: 36 W1, 37 Wo, 38 b1, 39 bo
  // ln0: 40 l.b, 41 l.g, 42 n.b, 43 n.g ; ln1: 44..47 ; ln2: 48..51

  // -------------------- workspace bump allocator --------------------
  char* wp = (char*)d_ws;
  auto alloc = [&](size_t bytes) -> void* {
    void* r = (void*)wp;
    wp += (bytes + 255) & ~(size_t)255;
    return r;
  };
  float* FS  = (float*)alloc((size_t)NL * HDIM * 4);
  float* FD  = (float*)alloc((size_t)NL * HDIM * 4);
  float* RST = (float*)alloc((size_t)NL * HDIM * 4);
  float* h0L = (float*)alloc((size_t)NL * HDIM * 4);
  float* h0N = (float*)alloc((size_t)NN * HDIM * 4);
  float* h1L = (float*)alloc((size_t)NL * HDIM * 4);
  float* h1N = (float*)alloc((size_t)NN * HDIM * 4);
  bf16_t* AL = (bf16_t*)alloc((size_t)MtL * KT_H * 512 * 2);
  bf16_t* AN = (bf16_t*)alloc((size_t)MtN * KT_H * 512 * 2);
  bf16_t* W0ls = (bf16_t*)alloc((size_t)64 * KT_L * 512 * 2);
  bf16_t* W0ld = (bf16_t*)alloc((size_t)64 * KT_N * 512 * 2);
  bf16_t* W0lr = (bf16_t*)alloc((size_t)64 * KT_N * 512 * 2);
  bf16_t* W0ns = (bf16_t*)alloc((size_t)64 * KT_N * 512 * 2);
  bf16_t* W0nd = (bf16_t*)alloc((size_t)64 * KT_L * 512 * 2);
  bf16_t* W0nr = (bf16_t*)alloc((size_t)64 * KT_L * 512 * 2);
  bf16_t* W1l  = (bf16_t*)alloc((size_t)64 * KT_H * 512 * 2);
  bf16_t* W1n  = (bf16_t*)alloc((size_t)64 * KT_H * 512 * 2);
  bf16_t* W2l  = (bf16_t*)alloc((size_t)64 * KT_H * 512 * 2);
  bf16_t* W2n  = (bf16_t*)alloc((size_t)64 * KT_H * 512 * 2);
  float* elb  = (float*)alloc((size_t)NL * 8 * 4);
  float* erb  = (float*)alloc((size_t)NL * 8 * 4);
  float* mb   = (float*)alloc((size_t)NL * 8 * 4);
  float* denb = (float*)alloc((size_t)NL * 8 * 4);
  float* eb   = (float*)alloc((size_t)E * 8 * 4);
  float* gstat = (float*)alloc((size_t)(2 * 1024 + 16) * 4);
  float* gsL = gstat, *gsN = gstat + 1024, *gcL = gstat + 2048, *gcN = gstat + 2056;

  // -------------------- helpers --------------------
  auto fill = [&](float* p, size_t n, float v) {
    k_fill<<<(unsigned)((n + 255) / 256), 256, 0, stream>>>(p, v, n);
  };
  auto convW = [&](const float* W, bf16_t* o, int Kreal, int KT) {
    size_t tot = (size_t)64 * KT * 512;
    k_conv_w<<<(unsigned)((tot + 255) / 256), 256, 0, stream>>>(W, o, Kreal, KT);
  };
  auto convA = [&](const float* X, bf16_t* o, int Kreal, int ld, int KT, int Mt) {
    size_t tot = (size_t)Mt * KT * 512;
    k_conv_act<<<(unsigned)((tot + 255) / 256), 256, 0, stream>>>(X, o, Kreal, ld, KT, tot);
  };
  auto gemm = [&](float* C, const bf16_t* A, const bf16_t* B, int Mt, int KT) {
    int waves = Mt * NGRP;
    k_gemm<<<(waves + 7) / 8, 256, 0, stream>>>(A, B, C, Mt, KT);
  };
  auto gat = [&](const bf16_t* Asrc, int KTs, const bf16_t* Ws, int MtS,
                 const bf16_t* Adst, int KTd, const bf16_t* Wd, int MtD,
                 const bf16_t* Wres, const float* resIdent,
                 const int* src, const int* dst, int Ndst,
                 const float* al, const float* ar, const float* bias,
                 const float* lg, const float* lb, float* hOut, bool last) {
    gemm(FS, Asrc, Ws, MtS, KTs);                  // fs = h_src @ Wsrc
    gemm(FD, Adst, Wd, MtD, KTd);                  // fd = h_dst @ Wdst
    int NHs = MtS * 16 * 8, NHd = MtD * 16 * 8;
    k_rowdot<<<NHs / 8, 256, 0, stream>>>(FS, al, elb, NHs);
    k_rowdot<<<NHd / 8, 256, 0, stream>>>(FD, ar, erb, NHd);
    const float* resPtr = resIdent;
    if (Wres) { gemm(FD, Adst, Wres, MtD, KTd); resPtr = FD; }  // FD reused
    fill(mb,   (size_t)Ndst * 8, -1e30f);
    fill(denb, (size_t)Ndst * 8, 0.f);
    fill(RST,  (size_t)Ndst * HDIM, 0.f);
    int E8 = E * 8;
    k_edge_logits<<<(E8 + 255) / 256, 256, 0, stream>>>(elb, erb, src, dst, eb, mb, E8);
    k_edge_exp<<<(E8 + 255) / 256, 256, 0, stream>>>(eb, mb, denb, dst, E8);
    k_edge_scatter<<<E, 256, 0, stream>>>(FS, eb, denb, src, dst, RST);
    if (!last) k_comb_ln<<<Ndst, 256, 0, stream>>>(RST, resPtr, bias, lg, lb, hOut);
    else       k_comb_mean_ln<<<Ndst, 128, 0, stream>>>(RST, resPtr, bias, lg, lb, hOut);
  };

  // -------------------- weight + layer-0 activation conversion --------------------
  convW(P[9],  W0ls, KL, KT_L);   convW(P[8],  W0ld, KN, KT_N);
  convW(P[13], W0lr, KN, KT_N);   convW(P[15], W0ns, KN, KT_N);
  convW(P[14], W0nd, KL, KT_L);   convW(P[19], W0nr, KL, KT_L);
  convW(P[20], W1l, HDIM, KT_H);  convW(P[24], W1n, HDIM, KT_H);
  convW(P[28], W2l, HDIM, KT_H);  convW(P[32], W2n, HDIM, KT_H);
  convA(feat_l, AL, KL, KL, KT_L, MtL);
  convA(feat_n, AN, KN, KN, KT_N, MtN);

  // -------------------- layer 0 (bipartite, GEMM residual) --------------------
  gat(AL, KT_L, W0ls, MtL, AN, KT_N, W0ld, MtN, W0lr, nullptr,
      l2n_src, l2n_dst, NN, P[10], P[11], P[12], P[43], P[42], h0N, false);
  gat(AN, KT_N, W0ns, MtN, AL, KT_L, W0nd, MtL, W0nr, nullptr,
      n2l_src, n2l_dst, NL, P[16], P[17], P[18], P[41], P[40], h0L, false);

  // -------------------- layer 1 (shared W, identity residual) --------------------
  convA(h0L, AL, HDIM, HDIM, KT_H, MtL);
  convA(h0N, AN, HDIM, HDIM, KT_H, MtN);
  gat(AL, KT_H, W1l, MtL, AN, KT_H, W1l, MtN, nullptr, h0N,
      l2n_src, l2n_dst, NN, P[21], P[22], P[23], P[47], P[46], h1N, false);
  gat(AN, KT_H, W1n, MtN, AL, KT_H, W1n, MtL, nullptr, h0L,
      n2l_src, n2l_dst, NL, P[25], P[26], P[27], P[45], P[44], h1L, false);

  // -------------------- layer 2 (head-mean output, dim 128) --------------------
  convA(h1L, AL, HDIM, HDIM, KT_H, MtL);
  convA(h1N, AN, HDIM, HDIM, KT_H, MtN);
  gat(AL, KT_H, W2l, MtL, AN, KT_H, W2l, MtN, nullptr, h1N,
      l2n_src, l2n_dst, NN, P[29], P[30], P[31], P[51], P[50], h0N, true);
  gat(AN, KT_H, W2n, MtN, AL, KT_H, W2n, MtL, nullptr, h1L,
      n2l_src, n2l_dst, NL, P[33], P[34], P[35], P[49], P[48], h0L, true);

  // -------------------- readout + MLP head --------------------
  fill(gstat, (size_t)(2 * 1024 + 16), 0.f);
  k_graph_accum<<<(NL * 128 + 255) / 256, 256, 0, stream>>>(h0L, gid_l, gsL, gcL, NL);
  k_graph_accum<<<(NN * 128 + 255) / 256, 256, 0, stream>>>(h0N, gid_n, gsN, gcN, NN);
  k_head<<<1, 256, 0, stream>>>(gsL, gcL, gsN, gcN, P[36], P[38], P[37], P[39],
                                (float*)d_out);
}
